// cross_scale_attention_12489764897427
// MI455X (gfx1250) — matrix operand order repaired and checked by reading, not compiled, not run on hardware
//
#include <hip/hip_runtime.h>

// ---------------------------------------------------------------------------
// Types for CDNA5 WMMA (wave32)
// ---------------------------------------------------------------------------
typedef __attribute__((ext_vector_type(16))) __bf16        v16bf;
typedef __attribute__((ext_vector_type(8)))  float         v8f;
typedef __attribute__((ext_vector_type(8)))  unsigned int  v8u;
typedef __attribute__((ext_vector_type(4)))  unsigned int  v4u;
typedef unsigned short u16;

__device__ __forceinline__ u16 f2bf(float f) {
  unsigned int x = __float_as_uint(f);
  unsigned int r = (x + 0x7fffu + ((x >> 16) & 1u)) >> 16;   // RNE
  return (u16)r;
}

__device__ __forceinline__ float leaky(float x) { return x > 0.f ? x : 0.01f * x; }

// Build a v16bf fragment from two 16B LDS reads (ds_load_b128 x2)
__device__ __forceinline__ v16bf frag_from(const u16* p0, const u16* p1) {
  v4u lo = *(const v4u*)p0;
  v4u hi = *(const v4u*)p1;
  v8u raw;
#pragma unroll
  for (int r = 0; r < 4; ++r) { raw[r] = lo[r]; raw[r + 4] = hi[r]; }
  return __builtin_bit_cast(v16bf, raw);
}

// A fragment (16x32 bf16, row-major LDS tile, stride 40 u16):
//   lane: row = base+(lane&15); needs K = half*8+[0..8) and 16+half*8+[0..8)
__device__ __forceinline__ v16bf frag_a(const u16* As, int row_base, int lane) {
  const int row  = row_base + (lane & 15);
  const int half = lane >> 4;
  const u16* p = As + row * 40 + half * 8;
  return frag_from(p, p + 16);
}

// B fragment (32x16 bf16, TRANSPOSED LDS tile Bst[col][k], stride 40 u16):
//   lane: col = base+(lane&15); needs K = half*8+[0..8) and 16+half*8+[0..8)
__device__ __forceinline__ v16bf frag_b(const u16* Bs, int col_base, int lane) {
  const int col  = col_base + (lane & 15);
  const int half = lane >> 4;
  const u16* p = Bs + col * 40 + half * 8;
  return frag_from(p, p + 16);
}

#define WMMA_BF16(A, B, C) \
  __builtin_amdgcn_wmma_f32_16x16x32_bf16(false, (A), false, (B), (short)0, (C), false, false)

// Async global -> LDS copy of 16B per lane (CDNA5 TDM-lite path, ASYNCcnt)
// LDS byte offset = low 32 bits of flat shared pointer (ISA 10.2: LDS aperture
// keeps workgroup-relative offset in addr[31:0]).
__device__ __forceinline__ void async_copy_b128(u16* lds_ptr, const u16* gaddr) {
  unsigned lds_off = (unsigned)(uintptr_t)(void*)lds_ptr;
  asm volatile("global_load_async_to_lds_b128 %0, %1, off"
               :: "v"(lds_off), "v"(gaddr)
               : "memory");
}

// ---------------------------------------------------------------------------
// Stage 1: 3x3 convs (+LeakyReLU)
// ---------------------------------------------------------------------------
__global__ void conv_q_kernel(const float* __restrict__ pan, const float* __restrict__ wq,
                              const float* __restrict__ bq, u16* __restrict__ qfea) {
  int idx = blockIdx.x * 256 + threadIdx.x;            // 2*64*384*384 exact
  int x = idx % 384; int t = idx / 384;
  int y = t % 384;   t /= 384;
  int c = t % 64;    int n = t / 64;
  const float* p = pan + (size_t)n * 384 * 384;
  float acc = bq[c];
#pragma unroll
  for (int dy = 0; dy < 3; ++dy)
#pragma unroll
    for (int dx = 0; dx < 3; ++dx) {
      int yy = y + dy - 1, xx = x + dx - 1;
      if ((unsigned)yy < 384u && (unsigned)xx < 384u)
        acc += p[yy * 384 + xx] * wq[(c * 3 + dy) * 3 + dx];
    }
  qfea[idx] = f2bf(leaky(acc));
}

__global__ void conv_k_kernel(const float* __restrict__ pan2, const float* __restrict__ wk,
                              const float* __restrict__ bk, float* __restrict__ kfea) {
  int idx = blockIdx.x * 256 + threadIdx.x;            // 2*64*96*96 exact
  int x = idx % 96; int t = idx / 96;
  int y = t % 96;   t /= 96;
  int c = t % 64;   int n = t / 64;
  const float* p = pan2 + (size_t)n * 96 * 96;
  float acc = bk[c];
#pragma unroll
  for (int dy = 0; dy < 3; ++dy)
#pragma unroll
    for (int dx = 0; dx < 3; ++dx) {
      int yy = y + dy - 1, xx = x + dx - 1;
      if ((unsigned)yy < 96u && (unsigned)xx < 96u)
        acc += p[yy * 96 + xx] * wk[(c * 3 + dy) * 3 + dx];
    }
  kfea[idx] = leaky(acc);
}

__global__ void conv_v_kernel(const float* __restrict__ ms, const float* __restrict__ wv,
                              const float* __restrict__ bv, float* __restrict__ vfea) {
  int idx = blockIdx.x * 256 + threadIdx.x;            // 2*8*96*96 exact
  int x = idx % 96; int t = idx / 96;
  int y = t % 96;   t /= 96;
  int c = t % 8;    int n = t / 8;
  const float* mp = ms + (size_t)n * 8 * 96 * 96;
  float acc = bv[c];
  for (int ci = 0; ci < 8; ++ci)
#pragma unroll
    for (int dy = 0; dy < 3; ++dy)
#pragma unroll
      for (int dx = 0; dx < 3; ++dx) {
        int yy = y + dy - 1, xx = x + dx - 1;
        if ((unsigned)yy < 96u && (unsigned)xx < 96u)
          acc += mp[(ci * 96 + yy) * 96 + xx] * wv[((c * 8 + ci) * 3 + dy) * 3 + dx];
      }
  vfea[idx] = leaky(acc);
}

// ---------------------------------------------------------------------------
// Rolled + SAME-padded sample: P[97][97] (pad bottom/right), rolled by (2,2)
// ---------------------------------------------------------------------------
__device__ __forceinline__ float sample_rolled(const float* f, int u, int v) {
  int yy = (u + 95) % 97;  // (u-2) mod 97
  int xx = (v + 95) % 97;
  return (yy < 96 && xx < 96) ? f[yy * 96 + xx] : 0.f;
}

__global__ __launch_bounds__(128) void knorm_kernel(const float* __restrict__ kfea,
                                                    float* __restrict__ norms) {
  __shared__ float red[128];
  int b = blockIdx.x;                       // 0..1151
  int n = b / 576, l = b % 576;
  int py = l / 24, px = l % 24;
  const float* kf = kfea + (size_t)n * 64 * 96 * 96;
  float acc = 0.f;
  for (int i = threadIdx.x; i < 1600; i += 128) {
    int c = i / 25, rem = i % 25, a = rem / 5, bb = rem % 5;
    float val = sample_rolled(kf + (size_t)c * 96 * 96, 4 * py + a, 4 * px + bb);
    acc += val * val;
  }
  red[threadIdx.x] = acc; __syncthreads();
  for (int off = 64; off > 0; off >>= 1) {
    if (threadIdx.x < off) red[threadIdx.x] += red[threadIdx.x + off];
    __syncthreads();
  }
  if (threadIdx.x == 0) norms[b] = sqrtf(red[0]);
}

__global__ __launch_bounds__(128) void kmax_kernel(const float* __restrict__ norms,
                                                   float* __restrict__ maxn) {
  __shared__ float red[128];
  int n = blockIdx.x;
  float mx = 0.f;
  for (int l = threadIdx.x; l < 576; l += 128) mx = fmaxf(mx, norms[n * 576 + l]);
  red[threadIdx.x] = mx; __syncthreads();
  for (int off = 64; off > 0; off >>= 1) {
    if (threadIdx.x < off) red[threadIdx.x] = fmaxf(red[threadIdx.x], red[threadIdx.x + off]);
    __syncthreads();
  }
  if (threadIdx.x == 0) maxn[n] = red[0];
}

// KmatT[n][l][kappa] (column-major operand: N=576 rows of K=1600)
__global__ void build_kmatT_kernel(const float* __restrict__ kfea, const float* __restrict__ maxn,
                                   u16* __restrict__ KmatT) {
  int idx = blockIdx.x * 256 + threadIdx.x;           // 2*576*1600 exact
  int kap = idx % 1600; int t = idx / 1600;
  int l = t % 576; int n = t / 576;
  int c = kap / 25, rem = kap % 25, a = rem / 5, b = rem % 5;
  int py = l / 24, px = l % 24;
  const float* kf = kfea + ((size_t)n * 64 + c) * 96 * 96;
  float val = sample_rolled(kf, 4 * py + a, 4 * px + b) / maxn[n];
  KmatT[idx] = f2bf(val);
}

// VmatT[n][col][l] (column-major operand: N=256(pad of 200) rows of K=576)
__global__ void build_vmatT_kernel(const float* __restrict__ vfea, u16* __restrict__ VmatT) {
  int idx = blockIdx.x * 256 + threadIdx.x;           // 2*256*576 exact
  int l = idx % 576; int t = idx / 576;
  int col = t % 256; int n = t / 256;
  float val = 0.f;
  if (col < 200) {
    int c = col / 25, rem = col % 25, a = rem / 5, b = rem % 5;
    int py = l / 24, px = l % 24;
    const float* vf = vfea + ((size_t)n * 8 + c) * 96 * 96;
    val = sample_rolled(vf, 4 * py + a, 4 * px + b);
  }
  VmatT[idx] = f2bf(val);
}

// Amat[n][m][kappa]: im2col of bf16 q features (zero SAME-pad folded in)
__global__ void build_amat_kernel(const u16* __restrict__ qfea, u16* __restrict__ Amat) {
  int idx = blockIdx.x * 256 + threadIdx.x;           // 2*9216*1600 exact
  int kap = idx % 1600; int t = idx / 1600;
  int m = t % 9216; int n = t / 9216;
  int oy = m / 96, ox = m % 96;
  int c = kap / 25, rem = kap % 25;
  int a = rem / 5, b = rem % 5;
  int y = 4 * oy + a, x = 4 * ox + b;
  u16 v = 0;
  if (y < 384 && x < 384)
    v = qfea[(((size_t)n * 64 + c) * 384 + y) * 384 + x];
  Amat[idx] = v;
}

// ---------------------------------------------------------------------------
// Unified bf16 WMMA GEMM: C[M x ldc] = A[M x K] * Bt[N x K]^T
// Block: 128 threads = 4 waves, tile 128(M) x 64(N); wave tile 64x32 (4x2 WMMA)
// Double-buffered LDS fed by global_load_async_to_lds_b128 (ASYNCcnt).
// ---------------------------------------------------------------------------
#define A_TILE (128 * 40)
#define B_TILE (64 * 40)

__device__ __forceinline__ void stage_tile_async(const u16* __restrict__ Ab,
                                                 const u16* __restrict__ Bb,
                                                 int K, int m0, int n0, int kap0,
                                                 u16* As, u16* Bs, int tid) {
#pragma unroll
  for (int j = 0; j < 4; ++j) {                 // 512 x 16B chunks of A
    int ci = tid + j * 128;
    int r = ci >> 2, c8 = (ci & 3) * 8;
    async_copy_b128(As + r * 40 + c8, Ab + (size_t)(m0 + r) * K + kap0 + c8);
  }
#pragma unroll
  for (int j = 0; j < 2; ++j) {                 // 256 x 16B chunks of B (K-contig)
    int ci = tid + j * 128;
    int col = ci >> 2, c8 = (ci & 3) * 8;
    async_copy_b128(Bs + col * 40 + c8, Bb + (size_t)(n0 + col) * K + kap0 + c8);
  }
}

__global__ __launch_bounds__(128) void gemm_bf16_kernel(const u16* __restrict__ A,
                                                        const u16* __restrict__ Bt,
                                                        float* __restrict__ C,
                                                        int K, int ldc,
                                                        size_t batchA, size_t batchB,
                                                        size_t batchC) {
  __shared__ __align__(16) u16 As[2 * A_TILE];   // row-major [row][k], double-buffered
  __shared__ __align__(16) u16 Bs[2 * B_TILE];   // transposed [col][k], double-buffered
  const int n  = blockIdx.z;
  const int m0 = blockIdx.x * 128;
  const int n0 = blockIdx.y * 64;
  const int tid = threadIdx.x, wave = tid >> 5, lane = tid & 31;
  const int wm = (wave >> 1) * 64, wn = (wave & 1) * 32;
  const u16* Ab = A  + (size_t)n * batchA;
  const u16* Bb = Bt + (size_t)n * batchB;

  v8f acc[4][2];
#pragma unroll
  for (int i = 0; i < 4; ++i)
#pragma unroll
    for (int j = 0; j < 2; ++j) acc[i][j] = (v8f){};

  const int kSteps = K >> 5;
  // prologue: stage tile 0 into buffer 0
  stage_tile_async(Ab, Bb, K, m0, n0, 0, As, Bs, tid);

  for (int kk = 0; kk < kSteps; ++kk) {
    u16* Asc = As + (kk & 1) * A_TILE;
    u16* Bsc = Bs + (kk & 1) * B_TILE;
    if (kk + 1 < kSteps) {
      // stream next tile into the other buffer while computing this one
      stage_tile_async(Ab, Bb, K, m0, n0, (kk + 1) * 32,
                       As + ((kk + 1) & 1) * A_TILE, Bs + ((kk + 1) & 1) * B_TILE, tid);
      // async loads complete in order: <=6 outstanding => this tile's 6 landed
      asm volatile("s_wait_asynccnt 0x6" ::: "memory");
    } else {
      asm volatile("s_wait_asynccnt 0x0" ::: "memory");
    }
    __syncthreads();
    v16bf b0 = frag_b(Bsc, wn, lane);
    v16bf b1 = frag_b(Bsc, wn + 16, lane);
#pragma unroll
    for (int i = 0; i < 4; ++i) {
      v16bf ai = frag_a(Asc, wm + i * 16, lane);
      acc[i][0] = WMMA_BF16(ai, b0, acc[i][0]);
      acc[i][1] = WMMA_BF16(ai, b1, acc[i][1]);
    }
    __syncthreads();   // protect buffer (kk&1) from next-next issue
  }

  float* Cb = C + (size_t)n * batchC;
  const int colL = lane & 15, halfL = lane >> 4;
#pragma unroll
  for (int i = 0; i < 4; ++i)
#pragma unroll
    for (int j = 0; j < 2; ++j)
#pragma unroll
      for (int v = 0; v < 8; ++v) {
        int row = m0 + wm + i * 16 + v + 8 * halfL;
        int col = n0 + wn + j * 16 + colL;
        Cb[(size_t)row * ldc + col] = acc[i][j][v];
      }
}

// ---------------------------------------------------------------------------
// Softmax over 576 patches per query position (x10 scale) -> bf16 attn
// ---------------------------------------------------------------------------
__global__ __launch_bounds__(128) void softmax_kernel(const float* __restrict__ scores,
                                                      u16* __restrict__ attn) {
  __shared__ float red[128];
  const int row = blockIdx.x;                  // 0 .. 2*9216-1
  const float* s = scores + (size_t)row * 576;
  u16* a = attn + (size_t)row * 576;
  const int tid = threadIdx.x;
  float mx = -3.4e38f;
  for (int l = tid; l < 576; l += 128) mx = fmaxf(mx, s[l]);
  red[tid] = mx; __syncthreads();
  for (int off = 64; off > 0; off >>= 1) {
    if (tid < off) red[tid] = fmaxf(red[tid], red[tid + off]);
    __syncthreads();
  }
  mx = red[0]; __syncthreads();
  float sum = 0.f;
  for (int l = tid; l < 576; l += 128) sum += __expf(10.f * (s[l] - mx));
  red[tid] = sum; __syncthreads();
  for (int off = 64; off > 0; off >>= 1) {
    if (tid < off) red[tid] += red[tid + off];
    __syncthreads();
  }
  float inv = 1.0f / red[0];
  for (int l = tid; l < 576; l += 128) a[l] = f2bf(__expf(10.f * (s[l] - mx)) * inv);
}

// ---------------------------------------------------------------------------
// Gather overlapping transpose-conv taps: res[n][c][y][x] = sum(U)/6
// ---------------------------------------------------------------------------
__global__ void gather_res_kernel(const float* __restrict__ U, float* __restrict__ res) {
  int idx = blockIdx.x * 256 + threadIdx.x;
  const int TOT = 2 * 8 * 385 * 385;
  if (idx >= TOT) return;
  int x = idx % 385; int t = idx / 385;
  int y = t % 385;   t /= 385;
  int c = t % 8;     int n = t / 8;
  const float* Un = U + (size_t)n * 9216 * 256;
  const int ry = y & 3, rx = x & 3;
  float s = 0.f;
#pragma unroll
  for (int ti = 0; ti < 2; ++ti) {
    int di = (ti == 0) ? ry : ry + 4;
    if (di > 4 || y < di) continue;
    int iy = (y - di) >> 2;
    if (iy >= 96) continue;
#pragma unroll
    for (int tj = 0; tj < 2; ++tj) {
      int dj = (tj == 0) ? rx : rx + 4;
      if (dj > 4 || x < dj) continue;
      int jx = (x - dj) >> 2;
      if (jx >= 96) continue;
      s += Un[(size_t)(iy * 96 + jx) * 256 + c * 25 + di * 5 + dj];
    }
  }
  res[idx] = s * (1.0f / 6.0f);
}

// ---------------------------------------------------------------------------
// Final 3x3 conv (+LeakyReLU) on res [2,8,385,385] -> d_out
// ---------------------------------------------------------------------------
__global__ void final_conv_kernel(const float* __restrict__ res, const float* __restrict__ wr,
                                  const float* __restrict__ br, float* __restrict__ out) {
  int idx = blockIdx.x * 256 + threadIdx.x;
  const int TOT = 2 * 8 * 385 * 385;
  if (idx >= TOT) return;
  int x = idx % 385; int t = idx / 385;
  int y = t % 385;   t /= 385;
  int c = t % 8;     int n = t / 8;
  const float* rp = res + (size_t)n * 8 * 385 * 385;
  float acc = br[c];
  for (int ci = 0; ci < 8; ++ci)
#pragma unroll
    for (int dy = 0; dy < 3; ++dy)
#pragma unroll
      for (int dx = 0; dx < 3; ++dx) {
        int yy = y + dy - 1, xx = x + dx - 1;
        if ((unsigned)yy < 385u && (unsigned)xx < 385u)
          acc += rp[((size_t)ci * 385 + yy) * 385 + xx] * wr[((c * 8 + ci) * 3 + dy) * 3 + dx];
      }
  out[idx] = leaky(acc);
}

// ---------------------------------------------------------------------------
// Launcher
// ---------------------------------------------------------------------------
extern "C" void kernel_launch(void* const* d_in, const int* in_sizes, int n_in,
                              void* d_out, int out_size, void* d_ws, size_t ws_size,
                              hipStream_t stream) {
  const float* ms   = (const float*)d_in[0];
  const float* pan  = (const float*)d_in[1];
  const float* pan2 = (const float*)d_in[2];
  const float* wq   = (const float*)d_in[3];
  const float* bq   = (const float*)d_in[4];
  const float* wk   = (const float*)d_in[5];
  const float* bk   = (const float*)d_in[6];
  const float* wv   = (const float*)d_in[7];
  const float* bv   = (const float*)d_in[8];
  const float* wr   = (const float*)d_in[9];
  const float* br   = (const float*)d_in[10];
  float* out = (float*)d_out;

  // Workspace carve-up (256B aligned) with region aliasing:
  //  regionA: Amat (59.0MB) ; after GEMM1: attn (21.2MB @0) + res (9.5MB @21.2MB)
  //  regionB: scores (42.5MB); after softmax: U (18.9MB @0)
  char* w = (char*)d_ws;
  auto carve = [&](size_t bytes) -> void* {
    void* p = (void*)w;
    w += (bytes + 255) & ~(size_t)255;
    return p;
  };
  u16*   qfea   = (u16*)  carve((size_t)2 * 64 * 384 * 384 * 2);
  float* kfea   = (float*)carve((size_t)2 * 64 * 96 * 96 * 4);
  float* vfea   = (float*)carve((size_t)2 * 8 * 96 * 96 * 4);
  float* norms  = (float*)carve((size_t)2 * 576 * 4);
  float* maxn   = (float*)carve(16);
  u16*   KmatT  = (u16*)  carve((size_t)2 * 576 * 1600 * 2);
  u16*   VmatT  = (u16*)  carve((size_t)2 * 256 * 576 * 2);
  char*  regionA = (char*)carve((size_t)2 * 9216 * 1600 * 2);   // 58.98 MB
  char*  regionB = (char*)carve((size_t)2 * 9216 * 576 * 4);    // 42.47 MB

  u16*   Amat   = (u16*)regionA;
  u16*   attn   = (u16*)regionA;                                // aliases dead Amat
  float* res    = (float*)(regionA + (size_t)2 * 9216 * 576 * 2);
  float* scores = (float*)regionB;
  float* U      = (float*)regionB;                              // aliases dead scores

  // 1) feature convs
  conv_q_kernel<<<(2 * 64 * 384 * 384) / 256, 256, 0, stream>>>(pan, wq, bq, qfea);
  conv_k_kernel<<<(2 * 64 * 96 * 96) / 256, 256, 0, stream>>>(pan2, wk, bk, kfea);
  conv_v_kernel<<<(2 * 8 * 96 * 96) / 256, 256, 0, stream>>>(ms, wv, bv, vfea);

  // 2) patch norms, global max, dense column-major K/V operands, im2col A
  knorm_kernel<<<2 * 576, 128, 0, stream>>>(kfea, norms);
  kmax_kernel<<<2, 128, 0, stream>>>(norms, maxn);
  build_kmatT_kernel<<<(2 * 576 * 1600) / 256, 256, 0, stream>>>(kfea, maxn, KmatT);
  build_vmatT_kernel<<<(2 * 256 * 576) / 256, 256, 0, stream>>>(vfea, VmatT);
  build_amat_kernel<<<(2 * 9216 * 1600) / 256, 256, 0, stream>>>(qfea, Amat);

  // 3) scores GEMM (WMMA bf16): M=9216, N=576, K=1600
  gemm_bf16_kernel<<<dim3(72, 9, 2), 128, 0, stream>>>(
      Amat, KmatT, scores, 1600, 576,
      (size_t)9216 * 1600, (size_t)576 * 1600, (size_t)9216 * 576);

  // 4) softmax over patches (writes attn over dead Amat region)
  softmax_kernel<<<2 * 9216, 128, 0, stream>>>(scores, attn);

  // 5) value GEMM (WMMA bf16): M=9216, N=256(pad of 200), K=576 (U over dead scores)
  gemm_bf16_kernel<<<dim3(72, 4, 2), 128, 0, stream>>>(
      attn, VmatT, U, 576, 256,
      (size_t)9216 * 576, (size_t)256 * 576, (size_t)9216 * 256);

  // 6) transpose-conv gather + /6
  gather_res_kernel<<<(2 * 8 * 385 * 385 + 255) / 256, 256, 0, stream>>>(U, res);

  // 7) final conv
  final_conv_kernel<<<(2 * 8 * 385 * 385 + 255) / 256, 256, 0, stream>>>(res, wr, br, out);
}
